// Model_39676907882339
// MI455X (gfx1250) — matrix operand order repaired and checked by reading, not compiled, not run on hardware
//
#include <hip/hip_runtime.h>
#include <hip/hip_bf16.h>

typedef _Float16 half_t;
typedef __attribute__((ext_vector_type(16))) _Float16 v16h;
typedef __attribute__((ext_vector_type(8)))  _Float16 v8h;
typedef __attribute__((ext_vector_type(8)))  float    v8f;
typedef __attribute__((ext_vector_type(4)))  float    v4f;

constexpr int   BATCH = 8;
constexpr int   L     = 2048;
constexpr int   D     = 128;
constexpr int   KB    = 64;                    // keys per block (softmax amortization)
constexpr float SCALE = 0.08838834764831845f;  // 1/sqrt(128)
constexpr float PDROP = 0.1f;
constexpr float KEEPS = 1.0f / 0.9f;           // folded into epilogue

// LDS row strides (in halfs), padded: 16B alignment for b128 loads + bank spread.
constexpr int KROW = 136;  // K tile [64 keys][128 d]
constexpr int VROW = 72;   // V^T tile [128 d][64 keys]
constexpr int PROW = 72;   // P tile per wave [16 rows][64 keys]

// Load 16 halfs (two 16B chunks) from LDS into a v16h.
__device__ inline v16h ld16(const half_t* p) {
    v8h lo = *(const v8h*)p;
    v8h hh = *(const v8h*)(p + 8);
    v16h r;
#pragma unroll
    for (int i = 0; i < 8; ++i) { r[i] = lo[i]; r[8 + i] = hh[i]; }
    return r;
}

// ---- DPP16-based 16-lane butterfly reductions (VALU, no LDS-pipe pressure).
template <int CTRL>
__device__ inline float dpp_mov(float x) {
    return __builtin_bit_cast(float,
        __builtin_amdgcn_mov_dpp(__builtin_bit_cast(int, x), CTRL, 0xF, 0xF, true));
}
__device__ inline float red_max16(float x) {
    x = fmaxf(x, dpp_mov<0xB1>(x));   // quad_perm(1,0,3,2)
    x = fmaxf(x, dpp_mov<0x4E>(x));   // quad_perm(2,3,0,1)
    x = fmaxf(x, dpp_mov<0x141>(x));  // row_half_mirror
    x = fmaxf(x, dpp_mov<0x140>(x));  // row_mirror
    return x;
}
__device__ inline float red_sum16(float x) {
    x += dpp_mov<0xB1>(x);
    x += dpp_mov<0x4E>(x);
    x += dpp_mov<0x141>(x);
    x += dpp_mov<0x140>(x);
    return x;
}

__global__ __launch_bounds__(128)
void fattn_f16wmma(const float* __restrict__ Q, const float* __restrict__ KV,
                   const float* __restrict__ QM, const float* __restrict__ U,
                   float* __restrict__ O)
{
    __shared__ half_t K_lds[KB * KROW];     // [key][d]  (B for S = Q K^T)
    __shared__ half_t Vt_lds[D * VROW];     // [d][key]  (B for O += P V)
    __shared__ half_t P_lds[4 * 16 * PROW]; // per-wave P tiles

    const int tid  = threadIdx.x;
    const int w    = tid >> 5;
    const int lane = tid & 31;
    const int m16  = lane & 15;
    const int hi   = lane >> 4;          // half-wave: selects contiguous K-range 16*hi..16*hi+15
    const int b    = blockIdx.y;
    const int qbase = blockIdx.x * 64 + w * 16;
    const int qrow  = qbase + m16;

    // ---- Q tile -> f16 A-matrix layout (16x32 per chunk, 4 chunks over D=128)
    v16h qa[4];
    {
        const float* qp = Q + (size_t)(b * L + qrow) * D;
#pragma unroll
        for (int c = 0; c < 4; ++c) {
            const int d0 = 32 * c + hi * 16;
            const int d1 = 32 * c + hi * 16 + 8;
            v4f f0 = *(const v4f*)(qp + d0);
            v4f f1 = *(const v4f*)(qp + d0 + 4);
            v4f f2 = *(const v4f*)(qp + d1);
            v4f f3 = *(const v4f*)(qp + d1 + 4);
#pragma unroll
            for (int i = 0; i < 4; ++i) {
                qa[c][i]      = (half_t)f0[i];
                qa[c][4 + i]  = (half_t)f1[i];
                qa[c][8 + i]  = (half_t)f2[i];
                qa[c][12 + i] = (half_t)f3[i];
            }
        }
    }

    // O accumulator: 16x128 fp32 in C-layout (8 N-tiles of v8f).
    v8f o[8];
#pragma unroll
    for (int t = 0; t < 8; ++t) o[t] = 0.0f;
    float mrow[8], lrow[8];
#pragma unroll
    for (int r = 0; r < 8; ++r) { mrow[r] = -3.0e38f; lrow[r] = 0.0f; }

    half_t* Pw = P_lds + w * 16 * PROW;

    // Staging pattern: lin = tid + 128*i -> row = tid>>5 + 4*i, c0 = (tid&31)*4.
    const int st_row0 = tid >> 5;
    const int st_c0   = (tid & 31) * 4;

    for (int kb = 0; kb < L; kb += KB) {
        __syncthreads();
        // ---- stage 64x128 of x2 into LDS as f16 (packed cvt), both layouts
        {
            const float* kp = KV + (size_t)(b * L + kb) * D;
#pragma unroll
            for (int i = 0; i < 16; ++i) {
                const int row = st_row0 + 4 * i;
                v4f f = *(const v4f*)(kp + row * D + st_c0);
                auto h01 = __builtin_amdgcn_cvt_pkrtz(f[0], f[1]);
                auto h23 = __builtin_amdgcn_cvt_pkrtz(f[2], f[3]);
                *(decltype(h01)*)&K_lds[row * KROW + st_c0]     = h01;
                *(decltype(h23)*)&K_lds[row * KROW + st_c0 + 2] = h23;
                Vt_lds[(st_c0 + 0) * VROW + row] = (half_t)h01[0];
                Vt_lds[(st_c0 + 1) * VROW + row] = (half_t)h01[1];
                Vt_lds[(st_c0 + 2) * VROW + row] = (half_t)h23[0];
                Vt_lds[(st_c0 + 3) * VROW + row] = (half_t)h23[1];
            }
        }
        __syncthreads();

        // ---- prefetch next K/V tile into cache (no VGPR cost, no LOADcnt)
        if (kb + KB < L) {
            const float* kpn = KV + (size_t)(b * L + kb + KB) * D;
#pragma unroll
            for (int i = 0; i < 16; ++i) {
                __builtin_prefetch(kpn + (st_row0 + 4 * i) * D + st_c0, 0, 3);
            }
        }

        // ---- S = Q K^T : four 16x16 tiles (keys kb+16j .. kb+16j+15)
        v8f s[4];
#pragma unroll
        for (int j = 0; j < 4; ++j) s[j] = 0.0f;
#pragma unroll
        for (int c = 0; c < 4; ++c) {
#pragma unroll
            for (int j = 0; j < 4; ++j) {
                v16h bb = ld16(&K_lds[(16 * j + m16) * KROW + 32 * c + hi * 16]);
                s[j] = __builtin_amdgcn_wmma_f32_16x16x32_f16(false, qa[c], false, bb,
                                                              (short)0, s[j], false, false);
            }
        }

        // ---- online softmax + mask + dropout-select; write P (f16) to LDS
#pragma unroll
        for (int r = 0; r < 8; ++r) {
            const int q = qbase + r + 8 * hi;            // C-layout row for VGPR r
            const size_t ub = (size_t)(b * L + q) * L + kb + m16;
            const size_t mb = (size_t)q * L + kb + m16;  // qm broadcast over batch
            float vv[4], uu[4];
#pragma unroll
            for (int j = 0; j < 4; ++j) {
                const float qmv = QM[mb + 16 * j];
                // u is streaming-only (134 MB): non-temporal, keep L2 for K/V
                uu[j] = __builtin_nontemporal_load(&U[ub + 16 * j]);
                vv[j] = s[j][r] * SCALE + qmv;
            }
            float mx = fmaxf(fmaxf(vv[0], vv[1]), fmaxf(vv[2], vv[3]));
            mx = red_max16(mx);                          // DPP16 butterfly (VALU)
            const float mn    = fmaxf(mrow[r], mx);
            const float alpha = __expf(mrow[r] - mn);
            float p[4];
#pragma unroll
            for (int j = 0; j < 4; ++j) p[j] = __expf(vv[j] - mn);
            const float rs = red_sum16((p[0] + p[1]) + (p[2] + p[3]));
            lrow[r] = lrow[r] * alpha + rs;   // normalizer includes all probs
            mrow[r] = mn;
#pragma unroll
            for (int t = 0; t < 8; ++t) o[t][r] *= alpha;
            const int prow = r + 8 * hi;
#pragma unroll
            for (int j = 0; j < 4; ++j) {
                // pure select; 1/0.9 scaling deferred to epilogue
                const float pj = (uu[j] >= PDROP) ? p[j] : 0.0f;
                Pw[prow * PROW + 16 * j + m16] = (half_t)pj;
            }
        }

        // ---- O += P V : P back in A-layout (two 16x32 chunks), V^T slices as B
        v16h pa0, pa1;
        {
            const half_t* pp = Pw + m16 * PROW + hi * 16;
            v8h a0 = *(const v8h*)pp;          // keys hi*16..+7
            v8h a1 = *(const v8h*)(pp + 8);    // keys hi*16+8..+15
            v8h a2 = *(const v8h*)(pp + 32);   // keys 32+hi*16..+7
            v8h a3 = *(const v8h*)(pp + 40);   // keys 40+hi*16..+7
#pragma unroll
            for (int i = 0; i < 8; ++i) {
                pa0[i] = a0[i]; pa0[8 + i] = a1[i];
                pa1[i] = a2[i]; pa1[8 + i] = a3[i];
            }
        }
#pragma unroll
        for (int t = 0; t < 8; ++t) {
            v16h vb0 = ld16(&Vt_lds[(16 * t + m16) * VROW + hi * 16]);
            o[t] = __builtin_amdgcn_wmma_f32_16x16x32_f16(false, pa0, false, vb0,
                                                          (short)0, o[t], false, false);
            v16h vb1 = ld16(&Vt_lds[(16 * t + m16) * VROW + 32 + hi * 16]);
            o[t] = __builtin_amdgcn_wmma_f32_16x16x32_f16(false, pa1, false, vb1,
                                                          (short)0, o[t], false, false);
        }
    }

    // ---- normalize (dropout keep-scale folded in) and store NT fp32 output
#pragma unroll
    for (int r = 0; r < 8; ++r) {
        const int q = qbase + r + 8 * hi;
        const float inv = KEEPS / lrow[r];
        const size_t ob = (size_t)(b * L + q) * D + m16;
#pragma unroll
        for (int t = 0; t < 8; ++t)
            __builtin_nontemporal_store(o[t][r] * inv, &O[ob + 16 * t]);
    }
}

extern "C" void kernel_launch(void* const* d_in, const int* in_sizes, int n_in,
                              void* d_out, int out_size, void* d_ws, size_t ws_size,
                              hipStream_t stream) {
    const float* Q  = (const float*)d_in[0];  // x1 [B,Lq,D]
    const float* KV = (const float*)d_in[1];  // x2 [B,Lk,D] (keys AND values)
    const float* QM = (const float*)d_in[2];  // qm [1,Lq,Lk]
    const float* U  = (const float*)d_in[3];  // u  [B,Lq,Lk]
    float* Out = (float*)d_out;

    dim3 grid(L / 64, BATCH);   // 64 query rows per workgroup
    dim3 block(128);            // 4 wave32s, 16 q-rows each
    hipLaunchKernelGGL(fattn_f16wmma, grid, block, 0, stream, Q, KV, QM, U, Out);
}
